// QuantizedDict_54812372631698
// MI455X (gfx1250) — matrix operand order repaired and checked by reading, not compiled, not run on hardware
//
#include <hip/hip_runtime.h>

typedef __attribute__((ext_vector_type(16))) __bf16 v16bf;
typedef __attribute__((ext_vector_type(8)))  __bf16 v8bf;
typedef __attribute__((ext_vector_type(8)))  float  v8f;

#define HS       256
#define K_CODES  4096
#define ROWS     64          // rows (spatial positions) per block; divides 1024 (h*w)
#define TC       64          // codebook tile held in LDS
#define XP       264         // padded LDS row pitch (bf16 elems) for x tile
#define EP       264         // padded LDS row pitch (bf16 elems) for emb tile
#define NOUT     8388608     // 32*256*32*32
#define INV_NUMEL (1.0f/8388608.0f)

// ---------------------------------------------------------------------------
// Prep: e_sq per code, bf16 hi/lo split of embedding, zero loss accumulators.
// ---------------------------------------------------------------------------
__global__ __launch_bounds__(256) void vq_prep_kernel(
    const float* __restrict__ emb, float* __restrict__ esq,
    __bf16* __restrict__ emb_h, __bf16* __restrict__ emb_l,
    float* __restrict__ loss) {
  const int code = blockIdx.x;
  const int t = threadIdx.x;
  float v = emb[(size_t)code * HS + t];
  __bf16 h = (__bf16)v;
  __bf16 l = (__bf16)(v - (float)h);
  emb_h[(size_t)code * HS + t] = h;
  emb_l[(size_t)code * HS + t] = l;
  float s = v * v;
#pragma unroll
  for (int off = 16; off >= 1; off >>= 1) s += __shfl_xor(s, off, 32);
  __shared__ float ws8[8];
  if ((t & 31) == 0) ws8[t >> 5] = s;
  __syncthreads();
  if (t == 0) {
    float tot = 0.f;
#pragma unroll
    for (int i = 0; i < 8; ++i) tot += ws8[i];
    esq[code] = tot;
    if (code == 0) { loss[0] = 0.f; loss[1] = 0.f; }
  }
}

// ---------------------------------------------------------------------------
// Main: per-block 64-row tile; WMMA bf16-split GEMM vs codebook; argmin;
// gather + NCHW write + loss. Two code-subtiles per pass share A registers.
// ---------------------------------------------------------------------------
__global__ __launch_bounds__(128) void vq_main_kernel(
    const float* __restrict__ x, const float* __restrict__ emb,
    const float* __restrict__ esq,
    const __bf16* __restrict__ emb_h, const __bf16* __restrict__ emb_l,
    float* __restrict__ out, float* __restrict__ loss) {

  // x-tile region (bf16 hi/lo), later reused as fp32 selected-rows buffer
  __shared__ alignas(16) unsigned char smem_x[2 * ROWS * XP * sizeof(__bf16)];
  __shared__ alignas(16) __bf16 es_h[TC * EP];
  __shared__ alignas(16) __bf16 es_l[TC * EP];
  __shared__ float esq_s[TC];
  __shared__ int   bidx_s[ROWS];
  __shared__ float lred[4];

  __bf16* xs_h = (__bf16*)smem_x;
  __bf16* xs_l = xs_h + ROWS * XP;
  float*  sel  = (float*)smem_x;          // [ROWS][257] after compute phase

  const int tid  = threadIdx.x;
  const int lane = tid & 31;
  const int wv   = tid >> 5;              // wave 0..3, each owns 16 rows
  const int m    = lane & 15;             // A row / B col / C col index
  const int half = lane >> 4;             // lane half selects K-subchunk / M+8

  const int n0  = blockIdx.x * ROWS;      // first flattened row (b*1024+hw)
  const int b   = n0 >> 10;
  const int hw0 = n0 & 1023;
  const size_t xbase = (size_t)b * HS * 1024 + (size_t)hw0;

  // ---- stage x tile: NCHW -> [row][chan] bf16 hi/lo in LDS ----
  for (int i = tid; i < ROWS * HS; i += 128) {
    int c = i >> 6;                       // channel
    int r = i & 63;                       // row in tile (contiguous hw)
    float v = x[xbase + (size_t)c * 1024 + (size_t)r];
    __bf16 h = (__bf16)v;
    __bf16 l = (__bf16)(v - (float)h);
    xs_h[r * XP + c] = h;
    xs_l[r * XP + c] = l;
  }

  float best[8];
  int   bidx[8];
#pragma unroll
  for (int v = 0; v < 8; ++v) { best[v] = 3.4e38f; bidx[v] = 0; }

  const int arow = wv * 16 + m;
  const __bf16* aph = xs_h + arow * XP;
  const __bf16* apl = xs_l + arow * XP;

  for (int ct = 0; ct < K_CODES; ct += TC) {
    __syncthreads();                      // prev compute done / x staging done
    // copy prepacked bf16 codebook tile into LDS (16B chunks, coalesced)
    {
      const uint4* gh = (const uint4*)(emb_h + (size_t)ct * HS);
      const uint4* gl = (const uint4*)(emb_l + (size_t)ct * HS);
      for (int i = tid; i < TC * (HS / 8); i += 128) {
        int row = i >> 5;
        int ch  = i & 31;
        uint4 dh = gh[row * 32 + ch];
        uint4 dl = gl[row * 32 + ch];
        *(uint4*)(es_h + row * EP + ch * 8) = dh;
        *(uint4*)(es_l + row * EP + ch * 8) = dl;
      }
      if (tid < TC) esq_s[tid] = esq[ct + tid];
    }
    __syncthreads();

#pragma unroll
    for (int sp = 0; sp < TC / 32; ++sp) {          // pair of 16-code subtiles
      const int s0 = 2 * sp, s1 = 2 * sp + 1;
      const __bf16* bph0 = es_h + (s0 * 16 + m) * EP;
      const __bf16* bpl0 = es_l + (s0 * 16 + m) * EP;
      const __bf16* bph1 = es_h + (s1 * 16 + m) * EP;
      const __bf16* bpl1 = es_l + (s1 * 16 + m) * EP;
      v8f acc0, acc1;
#pragma unroll
      for (int v = 0; v < 8; ++v) { acc0[v] = 0.f; acc1[v] = 0.f; }

#pragma unroll
      for (int k0 = 0; k0 < HS / 32; ++k0) {
        // A (16x32 bf16): lanes 0-15 K=k0*32+{0..7,16..23}; lanes 16-31 +8
        int ka = k0 * 32 + half * 8;
        // B (32x16 bf16): lanes 0-15 K=k0*32+{0..7,16..23}; lanes 16-31 +8
        int kb = k0 * 32 + half * 8;
        v8bf ah0 = *(const v8bf*)(aph + ka);
        v8bf ah1 = *(const v8bf*)(aph + ka + 16);
        v8bf al0 = *(const v8bf*)(apl + ka);
        v8bf al1 = *(const v8bf*)(apl + ka + 16);
        v16bf ah, al;
#pragma unroll
        for (int i = 0; i < 8; ++i) {
          ah[i] = ah0[i]; ah[i + 8] = ah1[i];
          al[i] = al0[i]; al[i + 8] = al1[i];
        }
        v8bf p0, p1;
        v16bf bh0v, bl0v, bh1v, bl1v;
        p0 = *(const v8bf*)(bph0 + kb); p1 = *(const v8bf*)(bph0 + kb + 16);
#pragma unroll
        for (int i = 0; i < 8; ++i) { bh0v[i] = p0[i]; bh0v[i + 8] = p1[i]; }
        p0 = *(const v8bf*)(bpl0 + kb); p1 = *(const v8bf*)(bpl0 + kb + 16);
#pragma unroll
        for (int i = 0; i < 8; ++i) { bl0v[i] = p0[i]; bl0v[i + 8] = p1[i]; }
        p0 = *(const v8bf*)(bph1 + kb); p1 = *(const v8bf*)(bph1 + kb + 16);
#pragma unroll
        for (int i = 0; i < 8; ++i) { bh1v[i] = p0[i]; bh1v[i + 8] = p1[i]; }
        p0 = *(const v8bf*)(bpl1 + kb); p1 = *(const v8bf*)(bpl1 + kb + 16);
#pragma unroll
        for (int i = 0; i < 8; ++i) { bl1v[i] = p0[i]; bl1v[i + 8] = p1[i]; }

        // 2-term split per subtile: x*e ~= ah*bh + ah*bl + al*bh (fp32 acc);
        // interleave the two subtiles for two independent WMMA dep-chains.
        acc0 = __builtin_amdgcn_wmma_f32_16x16x32_bf16(false, ah, false, bh0v,
                                                       (short)0, acc0, false, false);
        acc1 = __builtin_amdgcn_wmma_f32_16x16x32_bf16(false, ah, false, bh1v,
                                                       (short)0, acc1, false, false);
        acc0 = __builtin_amdgcn_wmma_f32_16x16x32_bf16(false, ah, false, bl0v,
                                                       (short)0, acc0, false, false);
        acc1 = __builtin_amdgcn_wmma_f32_16x16x32_bf16(false, ah, false, bl1v,
                                                       (short)0, acc1, false, false);
        acc0 = __builtin_amdgcn_wmma_f32_16x16x32_bf16(false, al, false, bh0v,
                                                       (short)0, acc0, false, false);
        acc1 = __builtin_amdgcn_wmma_f32_16x16x32_bf16(false, al, false, bh1v,
                                                       (short)0, acc1, false, false);
      }
      // d2 (up to per-row constant) = ||e||^2 - 2 x.e ; C layout: lane%16 = N,
      // vgpr v = M (lanes 0-15) or M-8 (lanes 16-31)
      float eq0 = esq_s[s0 * 16 + m];
      float eq1 = esq_s[s1 * 16 + m];
      int cg0 = ct + s0 * 16 + m;
      int cg1 = ct + s1 * 16 + m;
#pragma unroll
      for (int v = 0; v < 8; ++v) {
        float d0 = eq0 - 2.0f * acc0[v];
        float d1 = eq1 - 2.0f * acc1[v];
        if (d0 < best[v]) { best[v] = d0; bidx[v] = cg0; }
        if (d1 < best[v]) { best[v] = d1; bidx[v] = cg1; }
      }
    }
  }

  // cross-lane argmin: reduce over the 16 lanes of each half (same row set)
#pragma unroll
  for (int off = 8; off >= 1; off >>= 1) {
#pragma unroll
    for (int v = 0; v < 8; ++v) {
      float ov = __shfl_xor(best[v], off, 32);
      int   oi = __shfl_xor(bidx[v], off, 32);
      if (ov < best[v] || (ov == best[v] && oi < bidx[v])) {
        best[v] = ov; bidx[v] = oi;
      }
    }
  }
  if (m == 0) {                            // lanes 0 and 16
#pragma unroll
    for (int v = 0; v < 8; ++v) bidx_s[wv * 16 + half * 8 + v] = bidx[v];
  }
  __syncthreads();

  // gather selected fp32 codebook rows into LDS (coalesced row reads)
  for (int i = tid; i < ROWS * HS; i += 128) {
    int r = i >> 8;
    int c = i & 255;
    sel[r * 257 + c] = emb[(size_t)bidx_s[r] * HS + c];
  }
  __syncthreads();

  // NCHW write (64B-coalesced per half-wave) + loss accumulation
  float lacc = 0.f;
  for (int i = tid; i < ROWS * HS; i += 128) {
    int c = i >> 6;
    int r = i & 63;
    size_t g = xbase + (size_t)c * 1024 + (size_t)r;
    float ve = sel[r * 257 + c];
    float vx = x[g];
    float df = vx - ve;
    lacc += df * df;
    out[g] = ve;                           // oup == closest (straight-through)
  }
#pragma unroll
  for (int off = 16; off >= 1; off >>= 1) lacc += __shfl_xor(lacc, off, 32);
  if (lane == 0) lred[wv] = lacc;
  __syncthreads();
  if (tid == 0) {
    float tot = (lred[0] + lred[1] + lred[2] + lred[3]) * INV_NUMEL;
    atomicAdd(&loss[0], tot);              // dictionary_loss
    atomicAdd(&loss[1], tot);              // commitment_loss (same fwd value)
  }
}

// ---------------------------------------------------------------------------
extern "C" void kernel_launch(void* const* d_in, const int* in_sizes, int n_in,
                              void* d_out, int out_size, void* d_ws, size_t ws_size,
                              hipStream_t stream) {
  (void)in_sizes; (void)n_in; (void)out_size; (void)ws_size;
  const float* x   = (const float*)d_in[0];   // (32,256,32,32) fp32
  const float* emb = (const float*)d_in[1];   // (4096,256) fp32
  float* out  = (float*)d_out;                // [oup | dict_loss | commit_loss]
  float* loss = out + NOUT;

  // workspace layout: esq (16KB) | emb_h (2MB) | emb_l (2MB)
  float*  esq   = (float*)d_ws;
  __bf16* emb_h = (__bf16*)((char*)d_ws + (size_t)K_CODES * sizeof(float));
  __bf16* emb_l = emb_h + (size_t)K_CODES * HS;

  vq_prep_kernel<<<K_CODES, 256, 0, stream>>>(emb, esq, emb_h, emb_l, loss);
  vq_main_kernel<<<(NOUT / HS) / ROWS /*512*/, 128, 0, stream>>>(
      x, emb, esq, emb_h, emb_l, out, loss);
}
